// MultiHeadAttention_10857677324402
// MI455X (gfx1250) — matrix operand order repaired and checked by reading, compiled, not run on hardware
//
#include <hip/hip_runtime.h>

// MHA forward for B=4, T=2048, C=1024, H=16, D=64 on gfx1250 (MI455X).
// bf16 WMMA (v_wmma_f32_16x16x32_bf16, f32 accumulate) for every GEMM,
// fused flash-style causal attention (no T*T materialization).
// Round-3: attention K/V chunks staged with GLOBAL_LOAD_ASYNC_TO_LDS_B128
// (ASYNCcnt-tracked, double-buffered in LDS) so HBM/L2 latency overlaps
// WMMA + softmax of the current chunk. GEMM staging unchanged (needs 2B
// transpose, which DMA cannot do).
//
// Workspace layout (bytes), total ~92.3 MB:
//   xb    @ 0         : 8192x1024 bf16
//   wqb   @ 16777216  : 16x1024x64 bf16
//   wkb   @ 18874368  : 16x1024x64 bf16
//   wvb   @ 20971520  : 16x1024x64 bf16
//   wob   @ 23068672  : 1024x1024 bf16 (Wo transposed: wob[c][n] = Wo[n][c])
//   qb    @ 25165824  : [B,H,T,D] bf16
//   kb    @ 41943040  : [B,H,T,D] bf16
//   vT    @ 58720256  : [B,H,D,T] bf16
//   attnb @ 75497472  : 8192x1024 bf16

typedef __attribute__((ext_vector_type(16))) __bf16 v16bf;
typedef __attribute__((ext_vector_type(8)))  __bf16 v8bf;
typedef __attribute__((ext_vector_type(8)))  float  v8f;
typedef __attribute__((ext_vector_type(8)))  unsigned short v8u;

static __device__ inline unsigned short f2bf(float f) {
  unsigned int u = __builtin_bit_cast(unsigned int, f);
  u = (u + 0x7FFFu + ((u >> 16) & 1u)) >> 16;   // round-to-nearest-even
  return (unsigned short)u;
}

static __device__ inline v8f wmma_bf16(v16bf a, v16bf b, v8f c) {
  return __builtin_amdgcn_wmma_f32_16x16x32_bf16(false, a, false, b,
                                                 (short)0, c, false, false);
}

static __device__ inline v8f zero_v8f() {
  v8f z;
#pragma unroll
  for (int i = 0; i < 8; ++i) z[i] = 0.0f;
  return z;
}

// 16 contiguous bf16 (two aligned 16B halves at p, p+8)
static __device__ inline v16bf load_frag16(const unsigned short* p) {
  v8bf lo = *(const v8bf*)p;
  v8bf hi = *(const v8bf*)(p + 8);
  v16bf r;
#pragma unroll
  for (int j = 0; j < 8; ++j) { r[j] = lo[j]; r[8 + j] = hi[j]; }
  return r;
}

// A-layout fragment: elems 0..7 at p[0..7], elems 8..15 at p[16..23]
static __device__ inline v16bf load_frag_split(const unsigned short* p) {
  v8bf lo = *(const v8bf*)p;
  v8bf hi = *(const v8bf*)(p + 16);
  v16bf r;
#pragma unroll
  for (int j = 0; j < 8; ++j) { r[j] = lo[j]; r[8 + j] = hi[j]; }
  return r;
}

// Async DMA: global -> LDS, 16B per lane, tracked by ASYNCcnt.
// lds_off = low 32 bits of generic pointer (LDS aperture: addr[31:0]).
static __device__ inline void async_load_b128(unsigned lds_off, const void* g) {
  asm volatile("global_load_async_to_lds_b128 %0, %1, off"
               :: "v"(lds_off), "v"((unsigned long long)(size_t)g)
               : "memory");
}
#define S_WAIT_ASYNC(N) asm volatile("s_wait_asynccnt " #N ::: "memory")

// ---------------------------------------------------------------- convert ---
__global__ __launch_bounds__(256) void cvt_kernel(
    const float* __restrict__ x, const float* __restrict__ Wq,
    const float* __restrict__ Wk, const float* __restrict__ Wv,
    const float* __restrict__ Wo,
    unsigned short* xb, unsigned short* wqb, unsigned short* wkb,
    unsigned short* wvb, unsigned short* wob) {
  const long long XN = 8192LL * 1024, WN = 16LL * 1024 * 64, WON = 1024LL * 1024;
  const long long total = XN + 3 * WN + WON;
  for (long long i = blockIdx.x * (long long)blockDim.x + threadIdx.x; i < total;
       i += (long long)gridDim.x * blockDim.x) {
    if (i < XN) {
      xb[i] = f2bf(x[i]);
    } else if (i < XN + WN) {
      long long j = i - XN; wqb[j] = f2bf(Wq[j]);
    } else if (i < XN + 2 * WN) {
      long long j = i - XN - WN; wkb[j] = f2bf(Wk[j]);
    } else if (i < XN + 3 * WN) {
      long long j = i - XN - 2 * WN; wvb[j] = f2bf(Wv[j]);
    } else {
      long long j = i - XN - 3 * WN;          // wob[c][n] = Wo[n][c]
      long long c = j >> 10, n = j & 1023;
      wob[j] = f2bf(Wo[n * 1024 + c]);
    }
  }
}

// ------------------------------------------------------------------- QKV ----
__global__ __launch_bounds__(128) void qkv_kernel(
    const unsigned short* __restrict__ xb, const unsigned short* __restrict__ wqb,
    const unsigned short* __restrict__ wkb, const unsigned short* __restrict__ wvb,
    unsigned short* __restrict__ qb, unsigned short* __restrict__ kb,
    unsigned short* __restrict__ vT) {
  const int matId = blockIdx.x >> 7;      // 0..47 : proj*16 + head
  const int mblk  = blockIdx.x & 127;
  const int proj  = matId >> 4;           // 0=q 1=k 2=v
  const int h     = matId & 15;
  const int wave  = threadIdx.x >> 5, lane = threadIdx.x & 31;
  const int m = lane & 15, kgrp = lane >> 4;
  const int mbase = mblk * 64 + wave * 16;
  const unsigned short* W =
      (proj == 0 ? wqb : proj == 1 ? wkb : wvb) + (size_t)h * 65536;

  __shared__ alignas(16) unsigned short Wlds[64 * 40];  // [n][kk], padded

  v8f acc[4];
#pragma unroll
  for (int t = 0; t < 4; ++t) acc[t] = zero_v8f();

  for (int kbk = 0; kbk < 1024; kbk += 32) {
    __syncthreads();
    {  // coalesced b128 loads of the contiguous 32x64 tile, LDS-side transpose
      const v8u* src = (const v8u*)(W + (size_t)kbk * 64);
      for (int i = threadIdx.x; i < 256; i += 128) {
        v8u w = src[i];
        int kk = i >> 3, n0 = (i & 7) * 8;
#pragma unroll
        for (int j = 0; j < 8; ++j) Wlds[(n0 + j) * 40 + kk] = w[j];
      }
    }
    __syncthreads();
    const unsigned short* ar = xb + (size_t)(mbase + m) * 1024 + kbk + kgrp * 8;
    v16bf a = load_frag_split(ar);
#pragma unroll
    for (int nt = 0; nt < 4; ++nt) {
      v16bf b = load_frag_split(&Wlds[(nt * 16 + m) * 40 + kgrp * 8]);
      acc[nt] = wmma_bf16(a, b, acc[nt]);
    }
  }
#pragma unroll
  for (int nt = 0; nt < 4; ++nt) {
#pragma unroll
    for (int r = 0; r < 8; ++r) {
      int row = r + 8 * kgrp;
      int mrow = mbase + row;
      int bb = mrow >> 11, t = mrow & 2047;
      int d = nt * 16 + m;
      unsigned short val = f2bf(acc[nt][r]);
      size_t bh = (size_t)(bb * 16 + h);
      if (proj == 2) vT[(bh * 64 + d) * 2048 + t] = val;
      else (proj == 0 ? qb : kb)[(bh * 2048 + t) * 64 + d] = val;
    }
  }
}

// ------------------------------------------------------------- attention ----
// Double-buffered async K/V staging; Klds rows padded 64->72, VTlds/Plds 32->40.
__global__ __launch_bounds__(128) void attn_kernel(
    const unsigned short* __restrict__ qb, const unsigned short* __restrict__ kb,
    const unsigned short* __restrict__ vT, unsigned short* __restrict__ attnb) {
  const int bh   = blockIdx.x >> 5;   // b*16 + h
  const int qblk = blockIdx.x & 31;
  const int wave = threadIdx.x >> 5, lane = threadIdx.x & 31;
  const int m = lane & 15, kgrp = lane >> 4;
  const int qtb = qblk * 64 + wave * 16;
  const float scale = 0.125f;         // 1/sqrt(64)

  __shared__ alignas(16) unsigned short Klds[2][32 * 72];   // [key][d] padded
  __shared__ alignas(16) unsigned short VTlds[2][64 * 40];  // [d][key] padded
  __shared__ alignas(16) unsigned short Plds[4 * 16 * 40];

  const unsigned short* Ksrc = kb + (size_t)bh * 2048 * 64;
  const unsigned short* Vsrc = vT + (size_t)bh * 64 * 2048;

  // issue one 32-key chunk (4 async b128 per wave: 2 K + 2 V)
  auto issue_chunk = [&](int kbase, int buf) {
#pragma unroll
    for (int it = 0; it < 2; ++it) {
      int i = (int)threadIdx.x + it * 128;
      int key = i >> 3, part = i & 7;
      async_load_b128((unsigned)(size_t)&Klds[buf][key * 72 + part * 8],
                      Ksrc + (size_t)kbase * 64 + (size_t)i * 8);
    }
#pragma unroll
    for (int it = 0; it < 2; ++it) {
      int i = (int)threadIdx.x + it * 128;
      int d = i >> 2, part = i & 3;
      async_load_b128((unsigned)(size_t)&VTlds[buf][d * 40 + part * 8],
                      Vsrc + (size_t)d * 2048 + kbase + part * 8);
    }
  };

  const unsigned short* qrow = qb + ((size_t)bh * 2048 + qtb + m) * 64;
  v16bf qf[2];
#pragma unroll
  for (int f = 0; f < 2; ++f) qf[f] = load_frag16(qrow + f * 32 + kgrp * 16);

  float mrun[8], lrun[8];
#pragma unroll
  for (int r = 0; r < 8; ++r) { mrun[r] = -1e30f; lrun[r] = 0.0f; }
  v8f o[4];
#pragma unroll
  for (int t = 0; t < 4; ++t) o[t] = zero_v8f();

  const int nch = qblk * 2 + 2;
  issue_chunk(0, 0);

  for (int ch = 0; ch < nch; ++ch) {
    const int kbase = ch * 32;
    const int cur = ch & 1;
    if (ch + 1 < nch) {
      issue_chunk(kbase + 32, cur ^ 1);  // prefetch next chunk into other buffer
      S_WAIT_ASYNC(0x4);                 // drain this chunk's 4 transfers only
    } else {
      S_WAIT_ASYNC(0x0);
    }
    __syncthreads();                     // all waves' DMA results visible
    if (kbase <= qtb + 15) {             // wave-uniform causal skip
      v8f s0 = zero_v8f(), s1 = zero_v8f();
#pragma unroll
      for (int f = 0; f < 2; ++f) {
        v16bf b0 = load_frag16(&Klds[cur][m * 72 + f * 32 + kgrp * 16]);
        s0 = wmma_bf16(qf[f], b0, s0);
        v16bf b1 = load_frag16(&Klds[cur][(m + 16) * 72 + f * 32 + kgrp * 16]);
        s1 = wmma_bf16(qf[f], b1, s1);
      }
#pragma unroll
      for (int r = 0; r < 8; ++r) {
        int rowg = qtb + r + 8 * kgrp;
        int k0 = kbase + m, k1 = k0 + 16;
        float sa = (k0 <= rowg) ? s0[r] * scale : -1e30f;
        float sb = (k1 <= rowg) ? s1[r] * scale : -1e30f;
        float mx = fmaxf(sa, sb);
        mx = fmaxf(mx, __shfl_xor(mx, 1, 32));
        mx = fmaxf(mx, __shfl_xor(mx, 2, 32));
        mx = fmaxf(mx, __shfl_xor(mx, 4, 32));
        mx = fmaxf(mx, __shfl_xor(mx, 8, 32));
        float mn = fmaxf(mrun[r], mx);
        float alpha = __expf(mrun[r] - mn);
        float pa = __expf(sa - mn), pb = __expf(sb - mn);
        float sum = pa + pb;
        sum += __shfl_xor(sum, 1, 32);
        sum += __shfl_xor(sum, 2, 32);
        sum += __shfl_xor(sum, 4, 32);
        sum += __shfl_xor(sum, 8, 32);
        lrun[r] = lrun[r] * alpha + sum;
        mrun[r] = mn;
#pragma unroll
        for (int nt = 0; nt < 4; ++nt) o[nt][r] *= alpha;
        int row = r + 8 * kgrp;   // C->A relayout bounce via per-wave LDS
        Plds[wave * 640 + row * 40 + m] = f2bf(pa);
        Plds[wave * 640 + row * 40 + m + 16] = f2bf(pb);
      }
      __builtin_amdgcn_wave_barrier();  // intra-wave LDS store->load ordering
      v16bf pf = load_frag16(&Plds[wave * 640 + m * 40 + kgrp * 16]);
#pragma unroll
      for (int nt = 0; nt < 4; ++nt) {
        v16bf bv = load_frag16(&VTlds[cur][(nt * 16 + m) * 40 + kgrp * 16]);
        o[nt] = wmma_bf16(pf, bv, o[nt]);
      }
    }
    __syncthreads();   // end-of-compute: next iter may overwrite buf[cur]
  }
  const int bb = bh >> 4, h = bh & 15;
#pragma unroll
  for (int nt = 0; nt < 4; ++nt) {
#pragma unroll
    for (int r = 0; r < 8; ++r) {
      int row = r + 8 * kgrp;
      int t = qtb + row;
      float val = o[nt][r] / lrun[r];
      attnb[((size_t)(bb * 2048 + t)) * 1024 + h * 64 + nt * 16 + m] = f2bf(val);
    }
  }
}

// ------------------------------------------------------------- out proj -----
__global__ __launch_bounds__(128) void oproj_kernel(
    const unsigned short* __restrict__ attnb, const unsigned short* __restrict__ wob,
    const float* __restrict__ bo, float* __restrict__ out) {
  const int mblk = blockIdx.x & 127;
  const int nblk = blockIdx.x >> 7;
  const int wave = threadIdx.x >> 5, lane = threadIdx.x & 31;
  const int m = lane & 15, kgrp = lane >> 4;
  const int mbase = mblk * 64 + wave * 16;

  __shared__ alignas(16) unsigned short Wlds[64 * 40];  // [n][kk], padded

  v8f acc[4];
#pragma unroll
  for (int t = 0; t < 4; ++t) acc[t] = zero_v8f();

  for (int kbk = 0; kbk < 1024; kbk += 32) {
    __syncthreads();
    for (int i = threadIdx.x; i < 256; i += 128) {
      int kk = i >> 3, n0 = (i & 7) * 8;   // row-contiguous b128 load
      v8u w = *(const v8u*)(wob + (size_t)(kbk + kk) * 1024 + nblk * 64 + n0);
#pragma unroll
      for (int j = 0; j < 8; ++j) Wlds[(n0 + j) * 40 + kk] = w[j];
    }
    __syncthreads();
    const unsigned short* ar = attnb + (size_t)(mbase + m) * 1024 + kbk + kgrp * 8;
    v16bf a = load_frag_split(ar);
#pragma unroll
    for (int nt = 0; nt < 4; ++nt) {
      v16bf b = load_frag_split(&Wlds[(nt * 16 + m) * 40 + kgrp * 8]);
      acc[nt] = wmma_bf16(a, b, acc[nt]);
    }
  }
#pragma unroll
  for (int nt = 0; nt < 4; ++nt) {
#pragma unroll
    for (int r = 0; r < 8; ++r) {
      int row = r + 8 * kgrp;
      int col = nblk * 64 + nt * 16 + m;
      out[(size_t)(mbase + row) * 1024 + col] = acc[nt][r] + bo[col];
    }
  }
}

// ---------------------------------------------------------------------------
extern "C" void kernel_launch(void* const* d_in, const int* in_sizes, int n_in,
                              void* d_out, int out_size, void* d_ws, size_t ws_size,
                              hipStream_t stream) {
  const float* x  = (const float*)d_in[0];
  const float* Wq = (const float*)d_in[1];
  const float* Wk = (const float*)d_in[2];
  const float* Wv = (const float*)d_in[3];
  const float* Wo = (const float*)d_in[4];
  const float* bo = (const float*)d_in[5];
  float* out = (float*)d_out;

  char* ws = (char*)d_ws;
  unsigned short* xb    = (unsigned short*)(ws + 0);
  unsigned short* wqb   = (unsigned short*)(ws + 16777216);
  unsigned short* wkb   = (unsigned short*)(ws + 18874368);
  unsigned short* wvb   = (unsigned short*)(ws + 20971520);
  unsigned short* wob   = (unsigned short*)(ws + 23068672);
  unsigned short* qb    = (unsigned short*)(ws + 25165824);
  unsigned short* kbuf  = (unsigned short*)(ws + 41943040);
  unsigned short* vT    = (unsigned short*)(ws + 58720256);
  unsigned short* attnb = (unsigned short*)(ws + 75497472);

  cvt_kernel<<<4096, 256, 0, stream>>>(x, Wq, Wk, Wv, Wo, xb, wqb, wkb, wvb, wob);
  qkv_kernel<<<48 * 128, 128, 0, stream>>>(xb, wqb, wkb, wvb, qb, kbuf, vT);
  attn_kernel<<<2048, 128, 0, stream>>>(qb, kbuf, vT, attnb);
  oproj_kernel<<<2048, 128, 0, stream>>>(attnb, wob, bo, out);
}
